// VGATLayer_3453153706339
// MI455X (gfx1250) — compile-verified, matrix operand order checked by reading, not run on hardware
//
#include <hip/hip_runtime.h>
#include <hip/hip_bf16.h>
#include <math.h>

// ---------------------------------------------------------------------------
// GAT layer, sparse reformulation for MI455X (gfx1250, wave32, WMMA).
//
//  * h = inp @ W^T   : f16 WMMA 16x16x32, f32 accumulate. B (W strip) staged
//                      into LDS once per block via GLOBAL_LOAD_ASYNC_TO_LDS
//                      (ASYNCcnt) and shared by the 8 waves -> 8x less L2
//                      traffic, B fragments come from ds_load_b128.
//  * masked softmax  : only the N=12288 edges survive the -9e15 mask;
//                      rows with no edges -> uniform 1/N -> colmean(h).
//  * h' = attn @ h   : per-edge axpy with global f32 atomics (avg degree 1).
//  * out = elu(h')
//
// Traffic ~45 MB -> ~2 us at 23.3 TB/s; the dense reference path would move
// >600 MB. WMMA fragment layouts per CDNA5 ISA.
// ---------------------------------------------------------------------------

#define NN        12288
#define INF_      256
#define HID       256
#define NEGSLOPE  0.2f

typedef __attribute__((ext_vector_type(16))) _Float16 v16h;
typedef __attribute__((ext_vector_type(8)))  _Float16 v8h;
typedef __attribute__((ext_vector_type(8)))  float    v8f;

// ---- gfx1250 async copy global->LDS (ASYNCcnt-tracked, no VGPR round trip).
// LDS address: flat aperture keeps the LDS byte offset in addr[31:0], so
// truncating the generic pointer yields the DS address the HW expects.
__device__ inline void async_ld_b128(void* lds_dst, const void* gsrc)
{
    uint32_t loff  = (uint32_t)(uintptr_t)lds_dst;
    uint64_t gaddr = (uint64_t)(uintptr_t)gsrc;
    asm volatile("global_load_async_to_lds_b128 %0, %1, off"
                 :: "v"(loff), "v"(gaddr)
                 : "memory");
}

__device__ inline void wait_async0()
{
#if __has_builtin(__builtin_amdgcn_s_wait_asynccnt)
    __builtin_amdgcn_s_wait_asynccnt(0);
#else
    asm volatile("s_wait_asynccnt 0x0" ::: "memory");
#endif
}

// ---------------- prep: W = mu + softplus(rho)*eps  (to f16), inp -> f16 ----
__global__ void prep_kernel(const float* __restrict__ mu_w,
                            const float* __restrict__ rho_w,
                            const float* __restrict__ eps_w,
                            const float* __restrict__ inp,
                            _Float16* __restrict__ W16,
                            _Float16* __restrict__ inp16)
{
    const int idx = blockIdx.x * blockDim.x + threadIdx.x;
    if (idx < HID * INF_) {
        float r  = rho_w[idx];
        float sp = (r > 20.f) ? r : log1pf(__expf(r));   // softplus
        W16[idx] = (_Float16)(mu_w[idx] + sp * eps_w[idx]);
    }
    const int stride = gridDim.x * blockDim.x;
    for (int i = idx; i < NN * INF_; i += stride)
        inp16[i] = (_Float16)inp[i];
}

// ---------------- GEMM: h[NN][HID] = inp16 @ W16^T via WMMA -----------------
// Block = 8 waves sharing one 64-wide N strip; wave w -> M-tile m_group*8+w.
// B strip (64 x 256 f16 = 32 KB) staged in LDS via async copy.
__global__ __launch_bounds__(256) void gat_h_wmma(
        const _Float16* __restrict__ A16,   // [NN][INF_] row-major
        const _Float16* __restrict__ W16,   // [HID][INF_] row-major (= B K-major)
        float* __restrict__ h)              // [NN][HID]
{
    __shared__ _Float16 Bs[64 * INF_];      // 32 KB of the 320 KB/WGP LDS

    const int tid     = threadIdx.x;
    const int lane    = tid & 31;
    const int wave    = tid >> 5;
    const int n_strip = blockIdx.x & 3;     // 4 strips of 64 columns
    const int m_group = blockIdx.x >> 2;    // 0..95 -> 8 M-tiles each
    const int m_tile  = m_group * 8 + wave;
    const int r       = lane & 15;
    const int kgrp    = (lane >> 4) << 3;   // 0 or 8

    // ---- stage the 64x256 W strip into LDS: 256 thr x 8 async b128 each ----
    {
        const char* gsrc = (const char*)(W16 + (size_t)n_strip * 64 * INF_);
        char*       ldst = (char*)&Bs[0];
#pragma unroll
        for (int it = 0; it < 8; ++it) {
            const int byte_off = it * 4096 + tid * 16;
            async_ld_b128(ldst + byte_off, gsrc + byte_off);
        }
    }
    wait_async0();
    __syncthreads();

    const _Float16* arow = A16 + (size_t)(m_tile * 16 + r) * INF_ + kgrp;
    const _Float16* bl0  = &Bs[( 0 + r) * INF_ + kgrp];
    const _Float16* bl1  = &Bs[(16 + r) * INF_ + kgrp];
    const _Float16* bl2  = &Bs[(32 + r) * INF_ + kgrp];
    const _Float16* bl3  = &Bs[(48 + r) * INF_ + kgrp];

    v8f c0 = {}, c1 = {}, c2 = {}, c3 = {};

#pragma unroll
    for (int k0 = 0; k0 < INF_; k0 += 32) {
        // prefetch next K-slab of A (global_prefetch_b8; speculative OK)
        __builtin_prefetch(arow + k0 + 64, 0, 1);

        v8h alo = *(const v8h*)(arow + k0);
        v8h ahi = *(const v8h*)(arow + k0 + 16);
        v16h a  = __builtin_shufflevector(alo, ahi,
                    0,1,2,3,4,5,6,7,8,9,10,11,12,13,14,15);

        v8h l0 = *(const v8h*)(bl0 + k0), h0 = *(const v8h*)(bl0 + k0 + 16);
        v8h l1 = *(const v8h*)(bl1 + k0), h1 = *(const v8h*)(bl1 + k0 + 16);
        v8h l2 = *(const v8h*)(bl2 + k0), h2 = *(const v8h*)(bl2 + k0 + 16);
        v8h l3 = *(const v8h*)(bl3 + k0), h3 = *(const v8h*)(bl3 + k0 + 16);
        v16h bb0 = __builtin_shufflevector(l0, h0, 0,1,2,3,4,5,6,7,8,9,10,11,12,13,14,15);
        v16h bb1 = __builtin_shufflevector(l1, h1, 0,1,2,3,4,5,6,7,8,9,10,11,12,13,14,15);
        v16h bb2 = __builtin_shufflevector(l2, h2, 0,1,2,3,4,5,6,7,8,9,10,11,12,13,14,15);
        v16h bb3 = __builtin_shufflevector(l3, h3, 0,1,2,3,4,5,6,7,8,9,10,11,12,13,14,15);

        c0 = __builtin_amdgcn_wmma_f32_16x16x32_f16(false, a, false, bb0, (short)0, c0, false, false);
        c1 = __builtin_amdgcn_wmma_f32_16x16x32_f16(false, a, false, bb1, (short)0, c1, false, false);
        c2 = __builtin_amdgcn_wmma_f32_16x16x32_f16(false, a, false, bb2, (short)0, c2, false, false);
        c3 = __builtin_amdgcn_wmma_f32_16x16x32_f16(false, a, false, bb3, (short)0, c3, false, false);
    }

    // C/D layout: lanes 0-15 -> N=lane, M = vgpr (0..7); lanes 16-31 -> M = 8+vgpr
    float* hp = h + (size_t)(m_tile * 16 + (lane >> 4) * 8) * HID + n_strip * 64 + r;
#pragma unroll
    for (int v = 0; v < 8; ++v) {
        hp[(size_t)v * HID +  0] = c0[v];
        hp[(size_t)v * HID + 16] = c1[v];
        hp[(size_t)v * HID + 32] = c2[v];
        hp[(size_t)v * HID + 48] = c3[v];
    }
}

// ---------------- src = h@a1, dst = h@a2 (one row per wave) -----------------
__global__ void attn_coef_kernel(const float* __restrict__ h,
                                 const float* __restrict__ a_vec,
                                 float* __restrict__ src,
                                 float* __restrict__ dst)
{
    const int gwave = (blockIdx.x * blockDim.x + threadIdx.x) >> 5;
    const int lane  = threadIdx.x & 31;
    if (gwave >= NN) return;
    float s = 0.f, d = 0.f;
    for (int k = lane; k < HID; k += 32) {
        float hv = h[(size_t)gwave * HID + k];
        s += hv * a_vec[k];
        d += hv * a_vec[HID + k];
    }
#pragma unroll
    for (int off = 16; off; off >>= 1) {
        s += __shfl_xor(s, off, 32);
        d += __shfl_xor(d, off, 32);
    }
    if (lane == 0) { src[gwave] = s; dst[gwave] = d; }
}

// ---------------- init scratch (must re-zero every call) --------------------
__global__ void init_kernel(float* rowmax, float* rowsum, int* rowcnt,
                            float* hprime, float* colsum)
{
    const int idx = blockIdx.x * blockDim.x + threadIdx.x;
    if (idx < NN) { rowmax[idx] = -INFINITY; rowsum[idx] = 0.f; rowcnt[idx] = 0; }
    if (idx < HID) colsum[idx] = 0.f;
    const int stride = gridDim.x * blockDim.x;
    for (int i = idx; i < NN * HID; i += stride) hprime[i] = 0.f;
}

__device__ inline void atomicMaxF(float* addr, float value)
{
    int* ia  = (int*)addr;
    int  cur = *ia;
    while (__int_as_float(cur) < value) {
        int assumed = cur;
        cur = atomicCAS(ia, assumed, __float_as_int(value));
        if (cur == assumed) break;
    }
}

__device__ inline float leaky(float x) { return x > 0.f ? x : NEGSLOPE * x; }

// ---------------- edge pass 1: per-row max + degree -------------------------
__global__ void edge_max_kernel(const int* __restrict__ ei,
                                const float* __restrict__ src,
                                const float* __restrict__ dst,
                                float* rowmax, int* rowcnt)
{
    const int e = blockIdx.x * blockDim.x + threadIdx.x;
    if (e >= NN) return;
    const int i = ei[e], j = ei[NN + e];
    atomicMaxF(&rowmax[i], leaky(src[i] + dst[j]));
    atomicAdd(&rowcnt[i], 1);
}

// ---------------- edge pass 2: per-row exp-sum ------------------------------
__global__ void edge_sum_kernel(const int* __restrict__ ei,
                                const float* __restrict__ src,
                                const float* __restrict__ dst,
                                const float* __restrict__ rowmax,
                                float* rowsum)
{
    const int e = blockIdx.x * blockDim.x + threadIdx.x;
    if (e >= NN) return;
    const int i = ei[e], j = ei[NN + e];
    atomicAdd(&rowsum[i], __expf(leaky(src[i] + dst[j]) - rowmax[i]));
}

// ---------------- edge pass 3: h' += attn_ij * h[j] (one block per edge) ----
__global__ __launch_bounds__(HID) void edge_accum_kernel(
        const int* __restrict__ ei,
        const float* __restrict__ src,
        const float* __restrict__ dst,
        const float* __restrict__ rowmax,
        const float* __restrict__ rowsum,
        const float* __restrict__ h,
        float* __restrict__ hprime)
{
    const int e = blockIdx.x;
    const int i = ei[e], j = ei[NN + e];
    const float w = __expf(leaky(src[i] + dst[j]) - rowmax[i]) / rowsum[i];
    const int t = threadIdx.x;   // HID threads
    atomicAdd(&hprime[(size_t)i * HID + t], w * h[(size_t)j * HID + t]);
}

// ---------------- column sums of h (for empty-row uniform softmax) ----------
__global__ __launch_bounds__(HID) void colsum_kernel(const float* __restrict__ h,
                                                     float* __restrict__ colsum)
{
    const int t  = threadIdx.x;          // column
    const int r0 = blockIdx.x * 128;
    float s = 0.f;
    for (int r = r0; r < r0 + 128; ++r) s += h[(size_t)r * HID + t];
    atomicAdd(&colsum[t], s);
}

// ---------------- final: pick h' or colmean, apply ELU ----------------------
__global__ void final_kernel(const float* __restrict__ hprime,
                             const float* __restrict__ colsum,
                             const int* __restrict__ rowcnt,
                             float* __restrict__ out)
{
    const int idx = blockIdx.x * blockDim.x + threadIdx.x;   // NN*HID
    if (idx >= NN * HID) return;
    const int i = idx >> 8;          // / HID
    const int c = idx & 255;         // % HID
    float v = (rowcnt[i] > 0) ? hprime[idx] : colsum[c] * (1.0f / NN);
    out[idx] = v > 0.f ? v : (__expf(v) - 1.f);              // ELU, alpha = 1
}

// ---------------------------------------------------------------------------
extern "C" void kernel_launch(void* const* d_in, const int* in_sizes, int n_in,
                              void* d_out, int out_size, void* d_ws, size_t ws_size,
                              hipStream_t stream)
{
    const float* inp   = (const float*)d_in[0];
    const int*   ei    = (const int*)  d_in[1];
    const float* mu_w  = (const float*)d_in[2];
    const float* rho_w = (const float*)d_in[3];
    const float* eps_w = (const float*)d_in[4];
    const float* a_vec = (const float*)d_in[5];
    float* out = (float*)d_out;

    // workspace carve-up (~32 MB)
    char*  ws  = (char*)d_ws;
    size_t off = 0;
    auto carve = [&](size_t bytes) -> void* {
        void* p = ws + off;
        off = (off + bytes + 255) & ~(size_t)255;
        return p;
    };
    _Float16* inp16  = (_Float16*)carve((size_t)NN * INF_ * sizeof(_Float16));
    _Float16* W16    = (_Float16*)carve((size_t)HID * INF_ * sizeof(_Float16));
    float*    h      = (float*)   carve((size_t)NN * HID * sizeof(float));
    float*    hprime = (float*)   carve((size_t)NN * HID * sizeof(float));
    float*    src    = (float*)   carve((size_t)NN * sizeof(float));
    float*    dstv   = (float*)   carve((size_t)NN * sizeof(float));
    float*    rowmax = (float*)   carve((size_t)NN * sizeof(float));
    float*    rowsum = (float*)   carve((size_t)NN * sizeof(float));
    int*      rowcnt = (int*)     carve((size_t)NN * sizeof(int));
    float*    colsum = (float*)   carve((size_t)HID * sizeof(float));

    prep_kernel<<<3072, 256, 0, stream>>>(mu_w, rho_w, eps_w, inp, W16, inp16);
    init_kernel<<<3072, 256, 0, stream>>>(rowmax, rowsum, rowcnt, hprime, colsum);

    // 96 M-groups x 4 N-strips = 384 blocks, 8 waves each
    gat_h_wmma<<<384, 256, 0, stream>>>(inp16, W16, h);

    attn_coef_kernel<<<(NN * 32) / 256, 256, 0, stream>>>(h, a_vec, src, dstv);

    edge_max_kernel  <<<(NN + 255) / 256, 256, 0, stream>>>(ei, src, dstv, rowmax, rowcnt);
    edge_sum_kernel  <<<(NN + 255) / 256, 256, 0, stream>>>(ei, src, dstv, rowmax, rowsum);
    edge_accum_kernel<<<NN, HID, 0, stream>>>(ei, src, dstv, rowmax, rowsum, h, hprime);

    colsum_kernel<<<NN / 128, HID, 0, stream>>>(h, colsum);
    final_kernel<<<(NN * HID) / 256, 256, 0, stream>>>(hprime, colsum, rowcnt, out);
}
